// MultiHeadAttention_48206712930693
// MI455X (gfx1250) — matrix operand order repaired and checked by reading, not compiled, not run on hardware
//
#include <hip/hip_runtime.h>

// ---------------------------------------------------------------------------
// MHA forward for MI455X (gfx1250): bf16 WMMA + TDM (tensor_load_to_lds).
// B=2, S=2048, HIDDEN=1024, H=16, D=64.
// ---------------------------------------------------------------------------

typedef __attribute__((ext_vector_type(16))) __bf16   v16bf;
typedef __attribute__((ext_vector_type(8)))  float    v8f;
typedef __attribute__((ext_vector_type(4)))  unsigned v4u;
typedef __attribute__((ext_vector_type(8)))  unsigned v8u;

#define BATCH   2
#define SEQ     2048
#define HID     1024
#define NHEAD   16
#define HDIM    64
#define TOKENS  (BATCH * SEQ)   // 4096

union F16x16 { v16bf v; uint4 q[2]; };

__device__ __forceinline__ __bf16 f2bf(float f) {
  union { float f; unsigned u; } a; a.f = f;
  unsigned r = (a.u + 0x7fffu + ((a.u >> 16) & 1u)) >> 16;  // RNE
  union { unsigned short s; __bf16 b; } o; o.s = (unsigned short)r;
  return o.b;
}

__device__ __forceinline__ float half_max16(float v) {
  #pragma unroll
  for (int m = 1; m <= 8; m <<= 1) v = fmaxf(v, __shfl_xor(v, m, 32));
  return v;  // reduction across the 16 lanes of a half-wave
}
__device__ __forceinline__ float half_sum16(float v) {
  #pragma unroll
  for (int m = 1; m <= 8; m <<= 1) v += __shfl_xor(v, m, 32);
  return v;
}

// ---------------------------------------------------------------------------
// Tensor Data Mover: 2D tile descriptor (D# groups 0/1) + issue.
// group0: [1:0]=count(1), [63:32]=lds_addr, [120:64]=global_addr, [127:126]=type(2)
// group1: [17:16]=data_size(1 -> 2B), [79:48]=tensor_dim0, [111:80]=tensor_dim1,
//         [127:112]=tile_dim0, [143:128]=tile_dim1, [207:160]=tensor_dim0_stride
// ---------------------------------------------------------------------------
__device__ __forceinline__ v4u tdm_g0(const void* gptr, unsigned lds_off) {
  unsigned long long ga = (unsigned long long)(uintptr_t)gptr;
  v4u g;
  g[0] = 1u;                                   // count=1, user mode
  g[1] = lds_off;                              // LDS byte address
  g[2] = (unsigned)ga;                         // global_addr[31:0]
  g[3] = (unsigned)(ga >> 32) | (2u << 30);    // global_addr[56:32] | type=2
  return g;
}
__device__ __forceinline__ v8u tdm_g1_2d(unsigned dim0, unsigned dim1,
                                         unsigned tile0, unsigned tile1,
                                         unsigned stride0) {
  v8u g;
  g[0] = 1u << 16;                                   // data_size = 2 bytes
  g[1] = (dim0 & 0xFFFFu) << 16;                     // tensor_dim0[15:0]
  g[2] = (dim0 >> 16) | ((dim1 & 0xFFFFu) << 16);    // dim0[31:16] | dim1[15:0]
  g[3] = (dim1 >> 16) | (tile0 << 16);               // dim1[31:16] | tile_dim0
  g[4] = tile1 & 0xFFFFu;                            // tile_dim1 (tile_dim2=0)
  g[5] = stride0;                                    // tensor_dim0_stride[31:0]
  g[6] = 0u;                                         // stride hi / dim1_stride
  g[7] = 0u;
  return g;
}
__device__ __forceinline__ void tdm_issue(v4u g0, v8u g1) {
  // 2-group form: VADDR2/VADDR3 omitted (NULL) -> tensors up to 2D
  asm volatile("tensor_load_to_lds %0, %1" :: "s"(g0), "s"(g1) : "memory");
}

// ---------------------------------------------------------------------------
// fp32 -> bf16 elementwise convert
// ---------------------------------------------------------------------------
__global__ void f2bf_kernel(const float* __restrict__ in, __bf16* __restrict__ out, int n) {
  int i = blockIdx.x * blockDim.x + threadIdx.x;
  if (i < n) out[i] = f2bf(in[i]);
}

// ---------------------------------------------------------------------------
// Tiled GEMM: out[m,n] = sum_k A[m,k] * W[n,k] + bias[n]   (A @ W^T + b)
// A: [M,K] bf16 row-major, W: [N,K] bf16 row-major.
// mode 0: store bf16 into [B,H,S,D] layout (QKV projections)
// mode 1: store fp32 row-major [M,N] (final output projection)
// Block: 256 threads = 8 waves; tile 128(M) x 128(N), K-step 32.
// Staging: TDM double-buffered (wave 0 issues, TENSORcnt-gated), 1 barrier/step.
// Wave tile: 32(M) x 64(N) -> 2 A-frags x 4 B-frags = 8 WMMA per step.
// ---------------------------------------------------------------------------
__global__ __launch_bounds__(256) void gemm_bf16_kernel(
    const __bf16* __restrict__ A, const __bf16* __restrict__ W,
    const float* __restrict__ bias, __bf16* __restrict__ outq,
    float* __restrict__ outf, int mode, int M, int N, int K)
{
  __shared__ __bf16 lA[2][128 * 32];   // [m][k] double-buffered, 8 KiB each
  __shared__ __bf16 lB[2][128 * 32];   // [n][k]

  const int tid  = threadIdx.x;
  const int lane = tid & 31;
  const int w    = tid >> 5;
  const int lr   = lane & 15;
  const int kh   = lane >> 4;
  const int m0   = blockIdx.y * 128;
  const int n0   = blockIdx.x * 128;
  const int wm   = (w & 3) * 32;    // wave row base within tile
  const int wn   = (w >> 2) * 64;   // wave col base within tile
  const int NS   = K / 32;

  v8f acc[2][4] = {};

  const v8u g1A = tdm_g1_2d((unsigned)K, (unsigned)M, 32u, 128u, (unsigned)K);
  const v8u g1B = tdm_g1_2d((unsigned)K, (unsigned)N, 32u, 128u, (unsigned)K);
  const unsigned ldsA[2] = { (unsigned)(uintptr_t)&lA[0][0], (unsigned)(uintptr_t)&lA[1][0] };
  const unsigned ldsB[2] = { (unsigned)(uintptr_t)&lB[0][0], (unsigned)(uintptr_t)&lB[1][0] };

  if (w == 0) {
    tdm_issue(tdm_g0(A + (size_t)m0 * K, ldsA[0]), g1A);
    tdm_issue(tdm_g0(W + (size_t)n0 * K, ldsB[0]), g1B);
    __builtin_amdgcn_s_wait_tensorcnt(0);
  }
  __syncthreads();

  for (int s = 0; s < NS; ++s) {
    const int cur = s & 1;
    if (w == 0 && s + 1 < NS) {        // prefetch next K-slice into other buffer
      const int k1 = (s + 1) * 32;
      tdm_issue(tdm_g0(A + (size_t)m0 * K + k1, ldsA[cur ^ 1]), g1A);
      tdm_issue(tdm_g0(W + (size_t)n0 * K + k1, ldsB[cur ^ 1]), g1B);
    }

    F16x16 af[2];
    #pragma unroll
    for (int i = 0; i < 2; ++i) {
      const __bf16* arow = &lA[cur][(wm + i * 16 + lr) * 32];
      af[i].q[0] = *(const uint4*)(arow + 16 * kh);
      af[i].q[1] = *(const uint4*)(arow + 16 * kh + 8);
    }
    #pragma unroll
    for (int j = 0; j < 4; ++j) {
      F16x16 bf;
      const __bf16* brow = &lB[cur][(wn + j * 16 + lr) * 32 + 16 * kh];
      bf.q[0] = *(const uint4*)(brow);
      bf.q[1] = *(const uint4*)(brow + 8);
      #pragma unroll
      for (int i = 0; i < 2; ++i)
        acc[i][j] = __builtin_amdgcn_wmma_f32_16x16x32_bf16(
            false, af[i].v, false, bf.v, (short)0, acc[i][j], false, false);
    }

    if (w == 0 && s + 1 < NS) __builtin_amdgcn_s_wait_tensorcnt(0);
    __syncthreads();
  }

  // epilogue: bias + store (C layout: row = 8*kh + r, col = lr, per subtile)
  #pragma unroll
  for (int i = 0; i < 2; ++i) {
    #pragma unroll
    for (int j = 0; j < 4; ++j) {
      #pragma unroll
      for (int r = 0; r < 8; ++r) {
        const int row_g = m0 + wm + i * 16 + 8 * kh + r;
        const int col_g = n0 + wn + j * 16 + lr;
        const float val = acc[i][j][r] + bias[col_g];
        if (mode == 0) {
          const int h = col_g >> 6, d = col_g & 63;
          const int b = row_g >> 11, sq = row_g & 2047;  // S = 2048
          outq[(((size_t)(b * NHEAD + h) * SEQ) + sq) * HDIM + d] = f2bf(val);
        } else {
          outf[(size_t)row_g * N + col_g] = val;
        }
      }
    }
  }
}

// ---------------------------------------------------------------------------
// Flash attention (causal). Q,K,V: [B,H,S,D] bf16. O: [B,S,H*D] bf16.
// Block: 128 threads = 4 waves; grid (S/64, B*H).
// Each wave owns 16 query rows; streams key tiles of 64.
// ---------------------------------------------------------------------------
__global__ __launch_bounds__(128) void flash_attn_kernel(
    const __bf16* __restrict__ Q, const __bf16* __restrict__ K,
    const __bf16* __restrict__ V, __bf16* __restrict__ O)
{
  __shared__ __bf16 lK [64 * 64];  // [key][d]
  __shared__ __bf16 lVt[64 * 64];  // [d][key]  (transposed on stage)
  __shared__ __bf16 lP [64 * 64];  // [row][key] softmax probs, per-wave 16-row bands

  const int tid  = threadIdx.x;
  const int lane = tid & 31;
  const int w    = tid >> 5;
  const int lr   = lane & 15;
  const int kh   = lane >> 4;
  const int q0   = blockIdx.x * 64;
  const int bh   = blockIdx.y;          // b*NHEAD + h
  const size_t base = (size_t)bh * SEQ * HDIM;
  const __bf16* Qg = Q + base;
  const __bf16* Kg = K + base;
  const __bf16* Vg = V + base;

  // Load this wave's Q fragments once: rows q0+w*16+lr, two K=32 steps over D=64
  F16x16 aq[2];
  {
    const __bf16* qrow = Qg + (size_t)(q0 + w * 16 + lr) * HDIM;
    #pragma unroll
    for (int s2 = 0; s2 < 2; ++s2) {
      aq[s2].q[0] = *(const uint4*)(qrow + s2 * 32 + 16 * kh);
      aq[s2].q[1] = *(const uint4*)(qrow + s2 * 32 + 16 * kh + 8);
    }
  }

  float m_run[8], l_run[8];
  #pragma unroll
  for (int r = 0; r < 8; ++r) { m_run[r] = -__builtin_inff(); l_run[r] = 0.f; }
  v8f o_acc[4] = {};

  const int my_q    = q0 + w * 16;
  const float scale = 0.125f;           // 1/sqrt(64)

  for (int kt = 0; kt <= (int)blockIdx.x; ++kt) {
    const int kb = kt * 64;
    // stage K tile [key][d], V tile transposed to [d][key]
    {
      const int row  = tid >> 1;
      const int doff = (tid & 1) * 32;
      const uint4* sk = (const uint4*)(Kg + (size_t)(kb + row) * HDIM + doff);
      uint4* dk = (uint4*)(&lK[row * 64 + doff]);
      dk[0] = sk[0]; dk[1] = sk[1]; dk[2] = sk[2]; dk[3] = sk[3];
      union { uint4 q[4]; __bf16 h[32]; } tv;
      const uint4* sv = (const uint4*)(Vg + (size_t)(kb + row) * HDIM + doff);
      tv.q[0] = sv[0]; tv.q[1] = sv[1]; tv.q[2] = sv[2]; tv.q[3] = sv[3];
      #pragma unroll
      for (int j = 0; j < 32; ++j) lVt[(doff + j) * 64 + row] = tv.h[j];
    }
    __syncthreads();

    // scores S = Q @ K^T : 16x64 per wave, fp32 accum
    v8f s[4] = {};
    #pragma unroll
    for (int s2 = 0; s2 < 2; ++s2) {
      #pragma unroll
      for (int nb = 0; nb < 4; ++nb) {
        F16x16 bk;
        const __bf16* br = &lK[(nb * 16 + lr) * 64 + s2 * 32 + 16 * kh];
        bk.q[0] = *(const uint4*)(br);
        bk.q[1] = *(const uint4*)(br + 8);
        s[nb] = __builtin_amdgcn_wmma_f32_16x16x32_bf16(
            false, aq[s2].v, false, bk.v, (short)0, s[nb], false, false);
      }
    }

    // online softmax: each lane owns 8 rows (m = 8*kh + r), cols = lr per subtile
    float c[8];
    #pragma unroll
    for (int r = 0; r < 8; ++r) {
      const int qg = my_q + 8 * kh + r;
      float sv4[4];
      float rmax = -__builtin_inff();
      #pragma unroll
      for (int nb = 0; nb < 4; ++nb) {
        const int kg = kb + nb * 16 + lr;
        float v = s[nb][r] * scale;
        if (kg > qg) v = -__builtin_inff();   // causal mask
        sv4[nb] = v;
        rmax = fmaxf(rmax, v);
      }
      rmax = half_max16(rmax);
      const float mnew = fmaxf(m_run[r], rmax);
      float rsum = 0.f;
      #pragma unroll
      for (int nb = 0; nb < 4; ++nb) {
        const float p = __expf(sv4[nb] - mnew);
        rsum += p;
        lP[(w * 16 + 8 * kh + r) * 64 + nb * 16 + lr] = f2bf(p);
      }
      rsum = half_sum16(rsum);
      c[r] = __expf(m_run[r] - mnew);
      l_run[r] = l_run[r] * c[r] + rsum;
      m_run[r] = mnew;
    }
    #pragma unroll
    for (int nb = 0; nb < 4; ++nb)
      #pragma unroll
      for (int r = 0; r < 8; ++r)
        o_acc[nb][r] *= c[r];

    __syncthreads();   // lP visible / ordered before A-fragment reads

    // O += P @ V : K-dim = 64 keys -> two K=32 WMMA steps
    #pragma unroll
    for (int s2 = 0; s2 < 2; ++s2) {
      F16x16 ap;
      const __bf16* prow = &lP[(w * 16 + lr) * 64 + s2 * 32];
      ap.q[0] = *(const uint4*)(prow + 16 * kh);
      ap.q[1] = *(const uint4*)(prow + 16 * kh + 8);
      #pragma unroll
      for (int nb = 0; nb < 4; ++nb) {
        F16x16 bv;
        const __bf16* br = &lVt[(nb * 16 + lr) * 64 + s2 * 32 + 16 * kh];
        bv.q[0] = *(const uint4*)(br);
        bv.q[1] = *(const uint4*)(br + 8);
        o_acc[nb] = __builtin_amdgcn_wmma_f32_16x16x32_bf16(
            false, ap.v, false, bv.v, (short)0, o_acc[nb], false, false);
      }
    }
    __syncthreads();   // done with lK/lVt before next tile overwrites
  }

  // normalize & store to token-major [B, S, H*D] bf16
  const int b = bh >> 4, h = bh & 15;
  #pragma unroll
  for (int nb = 0; nb < 4; ++nb) {
    #pragma unroll
    for (int r = 0; r < 8; ++r) {
      const int qg  = my_q + 8 * kh + r;
      const int tok = b * SEQ + qg;
      const int col = h * HDIM + nb * 16 + lr;
      O[(size_t)tok * HID + col] = f2bf(o_acc[nb][r] / l_run[r]);
    }
  }
}

// ---------------------------------------------------------------------------
// Launch
// ---------------------------------------------------------------------------
extern "C" void kernel_launch(void* const* d_in, const int* in_sizes, int n_in,
                              void* d_out, int out_size, void* d_ws, size_t ws_size,
                              hipStream_t stream) {
  (void)in_sizes; (void)n_in; (void)out_size; (void)ws_size;
  const float* x  = (const float*)d_in[0];
  const float* Wq = (const float*)d_in[1];
  const float* bq = (const float*)d_in[2];
  const float* Wk = (const float*)d_in[3];
  const float* bk = (const float*)d_in[4];
  const float* Wv = (const float*)d_in[5];
  const float* bv = (const float*)d_in[6];
  const float* Wo = (const float*)d_in[7];
  const float* bo = (const float*)d_in[8];
  float* out = (float*)d_out;

  char* ws = (char*)d_ws;
  const size_t MB = 1u << 20;
  __bf16* d_xb   = (__bf16*)(ws);            //  8 MiB: x bf16 [4096,1024]
  __bf16* d_wqb  = (__bf16*)(ws +  8 * MB);  //  2 MiB each
  __bf16* d_wkb  = (__bf16*)(ws + 10 * MB);
  __bf16* d_wvb  = (__bf16*)(ws + 12 * MB);
  __bf16* d_wob  = (__bf16*)(ws + 14 * MB);
  __bf16* d_q    = (__bf16*)(ws + 16 * MB);  //  8 MiB: Q [B,H,S,D]
  __bf16* d_k    = (__bf16*)(ws + 24 * MB);  //  8 MiB
  __bf16* d_v    = (__bf16*)(ws + 32 * MB);  //  8 MiB
  __bf16* d_attn = (__bf16*)(ws + 40 * MB);  //  8 MiB: attn out [4096,1024]

  // fp32 -> bf16 conversions
  const int nx = TOKENS * HID;
  const int nw = HID * HID;
  f2bf_kernel<<<nx / 256, 256, 0, stream>>>(x,  d_xb,  nx);
  f2bf_kernel<<<nw / 256, 256, 0, stream>>>(Wq, d_wqb, nw);
  f2bf_kernel<<<nw / 256, 256, 0, stream>>>(Wk, d_wkb, nw);
  f2bf_kernel<<<nw / 256, 256, 0, stream>>>(Wv, d_wvb, nw);
  f2bf_kernel<<<nw / 256, 256, 0, stream>>>(Wo, d_wob, nw);

  // QKV projections: grid over 128x128 tiles
  dim3 gg(HID / 128, TOKENS / 128);   // (8, 32)
  gemm_bf16_kernel<<<gg, 256, 0, stream>>>(d_xb, d_wqb, bq, d_q, nullptr, 0, TOKENS, HID, HID);
  gemm_bf16_kernel<<<gg, 256, 0, stream>>>(d_xb, d_wkb, bk, d_k, nullptr, 0, TOKENS, HID, HID);
  gemm_bf16_kernel<<<gg, 256, 0, stream>>>(d_xb, d_wvb, bv, d_v, nullptr, 0, TOKENS, HID, HID);

  // causal flash attention
  dim3 fg(SEQ / 64, BATCH * NHEAD);   // (32, 32)
  flash_attn_kernel<<<fg, 128, 0, stream>>>(d_q, d_k, d_v, d_attn);

  // output projection (fp32 result)
  gemm_bf16_kernel<<<gg, 256, 0, stream>>>(d_attn, d_wob, bo, nullptr, out, 1, TOKENS, HID, HID);
}
